// Head_42082089566371
// MI455X (gfx1250) — matrix operand order repaired and checked by reading, not compiled, not run on hardware
//
#include <hip/hip_runtime.h>

// ---------------------------------------------------------------------------
// CDNA5 (gfx1250) fused attention head:
//   q,k,v = x@W*  ->  S = mask(q@k^T / sqrt(E))  ->  P = softmax(S)  ->  P@v
// All GEMMs: C = A (MxK, K-major) x B^T (B stored NxK, K-major), bf16 inputs,
// f32 WMMA accumulate.  Staging: global_load_async_to_lds_b128 (ASYNCcnt) with
// double-buffered LDS; each wave computes a 32x64 strip = 8 WMMAs per K-step.
// ---------------------------------------------------------------------------

typedef __attribute__((ext_vector_type(16))) __bf16        v16bf;
typedef __attribute__((ext_vector_type(8)))  float         v8f;
typedef __attribute__((ext_vector_type(4)))  unsigned int  u32x4;

union FragBF16 { u32x4 q[2]; v16bf v; };   // 32 bytes = 16 bf16 = one WMMA A/B frag

#define BM 128
#define BN 128
#define BK 32
#define NTHREADS 256

__device__ __forceinline__ unsigned short f2bf_rne(float f) {
  unsigned int u = __float_as_uint(f);
  u += 0x7fffu + ((u >> 16) & 1u);       // round-to-nearest-even
  return (unsigned short)(u >> 16);
}

// gfx1250 async global->LDS DMA (tracked by ASYNCcnt).
__device__ __forceinline__ void async_b128(unsigned lds_off, const void* gaddr) {
  asm volatile("global_load_async_to_lds_b128 %0, %1, off"
               :: "v"(lds_off), "v"((unsigned long long)(size_t)gaddr)
               : "memory");
}
__device__ __forceinline__ void wait_async0() {
  asm volatile("s_wait_asynccnt 0x0" ::: "memory");
}

// ---------------------------------------------------------------- conversions
__global__ __launch_bounds__(256)
void cvt_f32_bf16(const float* __restrict__ src, unsigned short* __restrict__ dst, int n) {
  int i = blockIdx.x * blockDim.x + threadIdx.x;
  if (i < n) dst[i] = f2bf_rne(src[i]);
}

// f32 [R][C] -> bf16 [C][R]   (one-time weight transpose; coalesced both sides)
__global__ __launch_bounds__(256)
void transpose_f32_bf16(const float* __restrict__ src, unsigned short* __restrict__ dst,
                        int R, int C) {
  __shared__ float tile[32][33];
  const int c0 = blockIdx.x * 32, r0 = blockIdx.y * 32;
  const int tx = threadIdx.x & 31, ty = threadIdx.x >> 5;   // 32 x 8
#pragma unroll
  for (int i = ty; i < 32; i += 8)
    tile[i][tx] = src[(size_t)(r0 + i) * C + (c0 + tx)];
  __syncthreads();
#pragma unroll
  for (int i = ty; i < 32; i += 8)
    dst[(size_t)(c0 + i) * R + (r0 + tx)] = f2bf_rne(tile[tx][i]);
}

// bf16 [R][C] -> bf16 [C][R]   (per-batch v transpose)
__global__ __launch_bounds__(256)
void transpose_bf16(const unsigned short* __restrict__ src, unsigned short* __restrict__ dst,
                    int R, int C) {
  __shared__ unsigned short tile[32][34];
  const int c0 = blockIdx.x * 32, r0 = blockIdx.y * 32;
  const int tx = threadIdx.x & 31, ty = threadIdx.x >> 5;
#pragma unroll
  for (int i = ty; i < 32; i += 8)
    tile[i][tx] = src[(size_t)(r0 + i) * C + (c0 + tx)];
  __syncthreads();
#pragma unroll
  for (int i = ty; i < 32; i += 8)
    dst[(size_t)(c0 + i) * R + (r0 + tx)] = tile[tx][i];
}

// ---------------------------------------------------------------- WMMA GEMM
// C[M,N] = A[M,K] @ Bt[N,K]^T.  Both operands bf16, K-contiguous.
//   MASK : causal(diagonal=+1) mask + scale, f32 out (scores)
//   OBF16: emit bf16 (projections) else f32
// Block: 128x128 tile, 8 waves (4 rows x 2 cols); wave strip = 32x64
// -> 2 A-frags x 4 B-frags = 8 WMMAs per 12 ds_load_b128 per K-step.
template<bool MASK, bool OBF16>
__global__ __launch_bounds__(NTHREADS)
void gemm_bf16_wmma(const unsigned short* __restrict__ A,
                    const unsigned short* __restrict__ Bt,
                    float* __restrict__ Cf,
                    unsigned short* __restrict__ Cb,
                    int M, int N, int K,
                    long long strideB, long long strideC,
                    float scale) {
  __shared__ __align__(16) unsigned short As[2][BM][BK];   // 16 KB
  __shared__ __align__(16) unsigned short Bs[2][BN][BK];   // 16 KB

  const int tid = threadIdx.x;
  const int m0  = blockIdx.y * BM;
  const int n0  = blockIdx.x * BN;
  const unsigned short* Bp = Bt + (size_t)blockIdx.z * (size_t)strideB;
  const long long cz = (long long)blockIdx.z * strideC;

  // Fully-masked tile: min s in tile > max t in tile + 1  ->  skip the GEMM.
  if (MASK && n0 > m0 + BM) {
    const u32x4 ninf = {0xFF800000u, 0xFF800000u, 0xFF800000u, 0xFF800000u};
    for (int i = tid; i < BM * BN / 4; i += NTHREADS) {
      const int r = i / (BN / 4);
      const int c = (i % (BN / 4)) * 4;
      *(u32x4*)&Cf[(size_t)(m0 + r) * N + (n0 + c)] = ninf;
    }
    return;
  }

  const int wid  = tid >> 5;
  const int lane = tid & 31;
  const int wm = (wid & 3) * 32;   // wave row offset (4 wave-rows x 32)
  const int wn = (wid >> 2) * 64;  // wave col offset (2 wave-cols x 64)
  const int lh = lane >> 4;        // lane half (0/1)
  const int lm = lane & 15;

  // per-thread staging coordinates: 2 b128 chunks for A, 2 for B
  const int sr = tid >> 1;                  // rows 0..127
  const int sc = (tid & 1) * 16;            // cols 0 / 16

  v8f acc[2][4];
#pragma unroll
  for (int i = 0; i < 2; ++i)
#pragma unroll
    for (int j = 0; j < 4; ++j) acc[i][j] = {};

  // -- issue DMA for K-tile k0 into buffer buf (4 x b128 per thread) --
  auto issue = [&](int buf, int k0) {
    async_b128((unsigned)(size_t)&As[buf][sr][sc],
               &A[(size_t)(m0 + sr) * K + (k0 + sc)]);
    async_b128((unsigned)(size_t)&As[buf][sr][sc + 8],
               &A[(size_t)(m0 + sr) * K + (k0 + sc + 8)]);
    async_b128((unsigned)(size_t)&Bs[buf][sr][sc],
               &Bp[(size_t)(n0 + sr) * K + (k0 + sc)]);
    async_b128((unsigned)(size_t)&Bs[buf][sr][sc + 8],
               &Bp[(size_t)(n0 + sr) * K + (k0 + sc + 8)]);
  };

  issue(0, 0);                                  // prologue

  int buf = 0;
  for (int k0 = 0; k0 < K; k0 += BK, buf ^= 1) {
    wait_async0();            // my DMA into buf is complete
    __syncthreads();          // everyone's DMA into buf complete;
                              // everyone done reading buf^1
    if (k0 + BK < K) issue(buf ^ 1, k0 + BK);   // overlap next tile's DMA

    // ---- fragments straight from LDS (two contiguous 16B chunks each) ----
    FragBF16 fa[2], fb[4];
#pragma unroll
    for (int i = 0; i < 2; ++i) {
      fa[i].q[0] = *(const u32x4*)&As[buf][wm + i * 16 + lm][8 * lh];
      fa[i].q[1] = *(const u32x4*)&As[buf][wm + i * 16 + lm][16 + 8 * lh];
    }
#pragma unroll
    for (int j = 0; j < 4; ++j) {
      fb[j].q[0] = *(const u32x4*)&Bs[buf][wn + j * 16 + lm][8 * lh];
      fb[j].q[1] = *(const u32x4*)&Bs[buf][wn + j * 16 + lm][16 + 8 * lh];
    }
#pragma unroll
    for (int i = 0; i < 2; ++i)
#pragma unroll
      for (int j = 0; j < 4; ++j)
        acc[i][j] = __builtin_amdgcn_wmma_f32_16x16x32_bf16(
            false, fa[i].v, false, fb[j].v, (short)0, acc[i][j], false, false);
  }

  // ---- store: C/D layout: VGPR r -> row (r + 8*half), col = lane&15 ----
#pragma unroll
  for (int i = 0; i < 2; ++i) {
#pragma unroll
    for (int j = 0; j < 4; ++j) {
      const int ng = n0 + wn + j * 16 + lm;
#pragma unroll
      for (int r = 0; r < 8; ++r) {
        const int mg = m0 + wm + i * 16 + r + 8 * lh;
        float val = acc[i][j][r];
        if (MASK) {
          val = (ng > mg + 1) ? -__builtin_inff() : val * scale;
          Cf[cz + (size_t)mg * N + ng] = val;
        } else if (OBF16) {
          Cb[cz + (size_t)mg * N + ng] = f2bf_rne(val);
        } else {
          Cf[cz + (size_t)mg * N + ng] = val;
        }
      }
    }
  }
}

// ---------------------------------------------------------------- softmax
__global__ __launch_bounds__(256)
void softmax_rows(const float* __restrict__ S, unsigned short* __restrict__ P, int N) {
  __shared__ float red[256];
  const int t   = blockIdx.x;
  const int tid = threadIdx.x;
  const float* row = S + (size_t)t * N;

  float mx = -__builtin_inff();
  for (int i = tid; i < N; i += 256) mx = fmaxf(mx, row[i]);
  red[tid] = mx; __syncthreads();
  for (int s = 128; s > 0; s >>= 1) {
    if (tid < s) red[tid] = fmaxf(red[tid], red[tid + s]);
    __syncthreads();
  }
  mx = red[0]; __syncthreads();

  float sum = 0.f;
  for (int i = tid; i < N; i += 256) sum += __expf(row[i] - mx);
  red[tid] = sum; __syncthreads();
  for (int s = 128; s > 0; s >>= 1) {
    if (tid < s) red[tid] += red[tid + s];
    __syncthreads();
  }
  const float inv = 1.0f / red[0];

  unsigned short* prow = P + (size_t)t * N;
  for (int i = tid; i < N; i += 256) prow[i] = f2bf_rne(__expf(row[i] - mx) * inv);
}

// ---------------------------------------------------------------- launcher
extern "C" void kernel_launch(void* const* d_in, const int* in_sizes, int n_in,
                              void* d_out, int out_size, void* d_ws, size_t ws_size,
                              hipStream_t stream) {
  constexpr int Bb = 8, T = 2048, E = 1024, H = 2048;

  const float* x  = (const float*)d_in[0];
  const float* Wq = (const float*)d_in[1];
  const float* Wk = (const float*)d_in[2];
  const float* Wv = (const float*)d_in[3];
  float* out = (float*)d_out;

  // ws layout (all K-major bf16 operands; total ~100 MB):
  //   wt  : 3*H*E bf16 (transposed weights, 12 MB)
  //   xb  : B*T*E bf16 (32 MB)
  //   qkv : 3*T*H bf16 (24 MB, per-batch)   vt : H*T bf16 (8 MB, per-batch)
  //   S   : T*T f32 (16 MB)                 P  : T*T bf16 (8 MB)
  unsigned short* wt  = (unsigned short*)d_ws;
  unsigned short* xb  = wt  + (size_t)3 * H * E;
  unsigned short* qkv = xb  + (size_t)Bb * T * E;
  unsigned short* vt  = qkv + (size_t)3 * T * H;
  float*          S   = (float*)(vt + (size_t)T * H);
  unsigned short* P   = (unsigned short*)(S + (size_t)T * T);

  // one-time: x -> bf16 ; W -> bf16 transposed (so every GEMM B operand is [N][K])
  cvt_f32_bf16<<<(Bb * T * E + 255) / 256, 256, 0, stream>>>(x, xb, Bb * T * E);
  transpose_f32_bf16<<<dim3(H / 32, E / 32), 256, 0, stream>>>(Wq, wt,                     E, H);
  transpose_f32_bf16<<<dim3(H / 32, E / 32), 256, 0, stream>>>(Wk, wt + (size_t)H * E,     E, H);
  transpose_f32_bf16<<<dim3(H / 32, E / 32), 256, 0, stream>>>(Wv, wt + (size_t)2 * H * E, E, H);

  const float inv_sqrt_e = 0.03125f;  // 1/sqrt(1024): reference scales by sqrt(E)

  for (int b = 0; b < Bb; ++b) {
    const unsigned short* xb_b = xb + (size_t)b * T * E;
    const unsigned short* q = qkv;
    const unsigned short* k = qkv + (size_t)T * H;
    const unsigned short* v = qkv + (size_t)2 * T * H;

    // q,k,v = x_b @ W*   (Bt = Wt[z] is [H][E]; grid.z selects weight/output slab)
    gemm_bf16_wmma<false, true>
        <<<dim3(H / BN, T / BM, 3), NTHREADS, 0, stream>>>(
            xb_b, wt, nullptr, qkv, T, H, E,
            (long long)H * E, (long long)T * H, 1.0f);

    // vt = v^T  (so P@v also uses the K-major fast path)
    transpose_bf16<<<dim3(H / 32, T / 32), 256, 0, stream>>>(v, vt, T, H);

    // S = mask(q @ k^T * 1/sqrt(E))   [Bt = k is [T][H] already]
    gemm_bf16_wmma<true, false>
        <<<dim3(T / BN, T / BM, 1), NTHREADS, 0, stream>>>(
            q, k, S, nullptr, T, T, H, 0, 0, inv_sqrt_e);

    // P = softmax(S) rows, bf16
    softmax_rows<<<T, 256, 0, stream>>>(S, P, T);

    // out_b = P @ v = P @ vt^T   (f32 straight to d_out)
    gemm_bf16_wmma<false, false>
        <<<dim3(H / BN, T / BM, 1), NTHREADS, 0, stream>>>(
            P, vt, out + (size_t)b * T * H, nullptr, T, H, T, 0, 0, 1.0f);
  }
}
